// AdvancedCRSN_77970836292121
// MI455X (gfx1250) — matrix operand order repaired and checked by reading, not compiled, not run on hardware
//
#include <hip/hip_runtime.h>
#include <hip/hip_bf16.h>

#define DEPTH 8
#define BATCH 131072

typedef __attribute__((ext_vector_type(16))) __bf16 v16bf;
typedef __attribute__((ext_vector_type(8)))  __bf16 v8bf;
typedef __attribute__((ext_vector_type(8)))  float  v8f;

union BF16x16 { v16bf v; v8bf h[2]; };

// 16 contiguous bf16 (B operand slice): two b128 LDS loads
__device__ __forceinline__ v16bf ld_contig16(const __bf16* p) {
  BF16x16 u;
  u.h[0] = *(const v8bf*)(p);
  u.h[1] = *(const v8bf*)(p + 8);
  return u.v;
}
// A operand slice: 16 contiguous bf16 at p (k..k+15)
__device__ __forceinline__ v16bf ld_a16(const __bf16* p) {
  BF16x16 u;
  u.h[0] = *(const v8bf*)(p);
  u.h[1] = *(const v8bf*)(p + 8);
  return u.v;
}
// sum across the 16 lanes of each half-wave (rows live per-half)
__device__ __forceinline__ float rsum16(float v) {
  #pragma unroll
  for (int m = 1; m < 16; m <<= 1) v += __shfl_xor(v, m, 32);
  return v;
}

// ---------------- prep: build bf16 weight tables in d_ws ----------------
// ws dword layout: [0,8192) WcT bf16[128][128]; [8192,10240) CbT bf16[32][128];
// [10240,12288) DwT bf16[32][128]; [12288,12320) cb_sq f32[32]; [12320,13344) sadj f32[32][32]
__global__ __launch_bounds__(256)
void crsn_prep(const float* __restrict__ Wr, const float* __restrict__ Wi,
               const float* __restrict__ codebook, const float* __restrict__ adj,
               const float* __restrict__ dec_w, float* __restrict__ out_scalars,
               void* __restrict__ ws)
{
  __bf16* WcT = (__bf16*)ws;            // [128][128], row n, col k
  __bf16* CbT = WcT + 128 * 128;        // [32][128]
  __bf16* DwT = CbT + 32 * 128;         // [32][128]
  float*  cbsq = (float*)(DwT + 32 * 128);
  float*  sadj = cbsq + 32;
  const int tid = threadIdx.x;

  for (int idx = tid; idx < 128 * 128; idx += 256) {
    int n = idx >> 7, k = idx & 127;
    float v;
    if (n < 64) v = (k < 64) ? Wr[n * 64 + k] : -Wi[n * 64 + (k - 64)];
    else        v = (k < 64) ? Wi[(n - 64) * 64 + k] : Wr[(n - 64) * 64 + (k - 64)];
    WcT[idx] = (__bf16)v;
  }
  for (int idx = tid; idx < 32 * 128; idx += 256) {
    CbT[idx] = (__bf16)codebook[idx];
    DwT[idx] = (__bf16)dec_w[idx];
  }
  for (int s = tid; s < 32; s += 256) {
    float acc = 0.f;
    for (int k = 0; k < 128; ++k) { float c = codebook[s * 128 + k]; acc += c * c; }
    cbsq[s] = acc;
  }
  for (int idx = tid; idx < 32 * 32; idx += 256) {
    float a = adj[idx];
    sadj[idx] = 0.1f / (1.f + __expf(-a));
  }
  if (tid < 2) out_scalars[tid] = 0.f;
}

// ---------------- main kernel: 8 waves x 16 rows each ----------------
__global__ __launch_bounds__(256)
void crsn_main(const int*   __restrict__ x,
               const float* __restrict__ emb_mag,
               const float* __restrict__ emb_phase,
               const float* __restrict__ ln_scale,
               const float* __restrict__ ln_shift,
               const float* __restrict__ mr_bias,
               const float* __restrict__ halt_w,
               const float* __restrict__ halt_b,
               const float* __restrict__ dec_b,
               const void*  __restrict__ ws,
               float*       __restrict__ out)
{
  __shared__ __bf16 sWc[128 * 128];   // 32 KB  combined complex weight (B operand, [n][k])
  __shared__ __bf16 sCb[32 * 128];    //  8 KB  codebook (B operand + zq gather)
  __shared__ __bf16 sDw[32 * 128];    //  8 KB  decoder
  __shared__ float  sCbsq[32];
  __shared__ float  sSadj[32 * 32];   //  4 KB  0.1*sigmoid(adj)
  __shared__ __bf16 sA[8][16 * 128];  // 32 KB  per-wave A-operand bounce buffer

  const int tid  = threadIdx.x;
  const int lane = tid & 31;
  const int wave = tid >> 5;
  const int ln   = lane & 15;   // column within tile / row id for A
  const int h    = lane >> 4;   // half-wave selector

  // cooperative LDS fill from ws
  {
    const unsigned* w32 = (const unsigned*)ws;
    unsigned* p;
    p = (unsigned*)sWc;   for (int i = tid; i < 8192; i += 256) p[i] = w32[i];
    p = (unsigned*)sCb;   for (int i = tid; i < 2048; i += 256) p[i] = w32[8192 + i];
    p = (unsigned*)sDw;   for (int i = tid; i < 2048; i += 256) p[i] = w32[10240 + i];
    p = (unsigned*)sCbsq; for (int i = tid; i < 32;   i += 256) p[i] = w32[12288 + i];
    p = (unsigned*)sSadj; for (int i = tid; i < 1024; i += 256) p[i] = w32[12320 + i];
  }
  __syncthreads();

  __bf16* ascr = sA[wave];
  const int base = blockIdx.x * 128 + wave * 16;

  // per-lane broadcast params
  float lnsc[4], lnsh[4], mrb[4], hw[8], cbsqr[2], decb[2];
  #pragma unroll
  for (int t = 0; t < 4; ++t) {
    lnsc[t] = ln_scale[t * 16 + ln];
    lnsh[t] = ln_shift[t * 16 + ln];
    mrb[t]  = mr_bias [t * 16 + ln];
  }
  #pragma unroll
  for (int t = 0; t < 8; ++t) hw[t] = halt_w[t * 16 + ln];
  const float hb = halt_b[0];
  #pragma unroll
  for (int t = 0; t < 2; ++t) { cbsqr[t] = sCbsq[t * 16 + ln]; decb[t] = dec_b[t * 16 + ln]; }

  // state in WMMA C-layout: ST[t][v] = element (row v+8h, col t*16+ln); t<4: zr, t>=4: zi
  float ST[8][8], AT[8][8];
  float halting[8], remain[8];
  int   sym[8];
  #pragma unroll
  for (int v = 0; v < 8; ++v) { halting[v] = 0.f; remain[v] = 1.f; sym[v] = 0; }
  #pragma unroll
  for (int t = 0; t < 8; ++t)
    #pragma unroll
    for (int v = 0; v < 8; ++v) AT[t][v] = 0.f;

  // embedding init: z = emb_mag * exp(i*emb_phase)
  #pragma unroll
  for (int v = 0; v < 8; ++v) {
    const int row = base + v + 8 * h;
    const int xi  = x[row];
    #pragma unroll
    for (int t = 0; t < 4; ++t) {
      const int j = t * 16 + ln;
      float mg = emb_mag[xi * 64 + j];
      float ph = emb_phase[xi * 64 + j];
      float sn, cs;
      __sincosf(ph, &sn, &cs);
      ST[t][v]     = mg * cs;
      ST[4 + t][v] = mg * sn;
    }
  }

  float pacc = 0.f, vqacc = 0.f;

  #pragma unroll 1
  for (int step = 0; step < DEPTH; ++step) {
    // --- state -> A scratch (bf16) ---
    #pragma unroll
    for (int t = 0; t < 8; ++t)
      #pragma unroll
      for (int v = 0; v < 8; ++v)
        ascr[(v + 8 * h) * 128 + t * 16 + ln] = (__bf16)ST[t][v];

    v16bf Ach[4];
    #pragma unroll
    for (int kc = 0; kc < 4; ++kc)
      Ach[kc] = ld_a16(ascr + ln * 128 + kc * 32 + h * 16);

    // --- combined complex matmul: [zr zi] @ Wc -> [nr ni] (8 N-tiles x 4 K-chunks) ---
    v8f OT[8];
    #pragma unroll
    for (int tn = 0; tn < 8; ++tn) {
      v8f c = {0.f, 0.f, 0.f, 0.f, 0.f, 0.f, 0.f, 0.f};
      #pragma unroll
      for (int kc = 0; kc < 4; ++kc) {
        v16bf b = ld_contig16(sWc + (tn * 16 + ln) * 128 + kc * 32 + h * 16);
        c = __builtin_amdgcn_wmma_f32_16x16x32_bf16(false, Ach[kc], false, b,
                                                    (short)0, c, false, false);
      }
      OT[tn] = c;
    }

    // --- magnitude stats (mean/var over 64 dims, ddof=1) ---
    float srow[8], s2row[8];
    #pragma unroll
    for (int v = 0; v < 8; ++v) { srow[v] = 0.f; s2row[v] = 0.f; }
    #pragma unroll
    for (int t = 0; t < 4; ++t)
      #pragma unroll
      for (int v = 0; v < 8; ++v) {
        float nr = OT[t][v], ni = OT[4 + t][v];
        float mag = sqrtf(nr * nr + ni * ni) + 1e-6f;
        srow[v] += mag; s2row[v] += mag * mag;
      }
    float mean[8], rstd[8];
    #pragma unroll
    for (int v = 0; v < 8; ++v) {
      float s1 = rsum16(srow[v]);
      float s2 = rsum16(s2row[v]);
      float m  = s1 * (1.f / 64.f);
      float var = (s2 - s1 * m) * (1.f / 63.f);
      mean[v] = m;
      rstd[v] = rsqrtf(var + 1e-6f);
    }

    // --- mag-LN + modReLU -> feats (in place over OT) ---
    #pragma unroll
    for (int t = 0; t < 4; ++t)
      #pragma unroll
      for (int v = 0; v < 8; ++v) {
        float nr  = OT[t][v], ni = OT[4 + t][v];
        float raw = sqrtf(nr * nr + ni * ni);
        float mag = raw + 1e-6f;
        float mn  = (mag - mean[v]) * rstd[v] * lnsc[t] + lnsh[t];
        float inv = mn / (raw + 1e-30f);
        float zr2 = nr * inv, zi2 = ni * inv;
        float mag2 = sqrtf(zr2 * zr2 + zi2 * zi2) + 1e-6f;
        float sc   = fmaxf(mag2 + mrb[t], 0.f) / mag2;
        OT[t][v]     = zr2 * sc;
        OT[4 + t][v] = zi2 * sc;
      }

    // --- halting prob + ||feats||^2 ---
    float hd[8], fsq[8];
    #pragma unroll
    for (int v = 0; v < 8; ++v) { hd[v] = 0.f; fsq[v] = 0.f; }
    #pragma unroll
    for (int t = 0; t < 8; ++t)
      #pragma unroll
      for (int v = 0; v < 8; ++v) {
        float f = OT[t][v];
        hd[v]  += f * hw[t];
        fsq[v] += f * f;
      }
    float p[8];
    #pragma unroll
    for (int v = 0; v < 8; ++v) {
      float d = rsum16(hd[v]);
      fsq[v]  = rsum16(fsq[v]);
      p[v] = 1.f / (1.f + __expf(-(d + hb)));
    }

    // --- feats -> scratch, codebook WMMA (2 N-tiles of 16 codes) ---
    #pragma unroll
    for (int t = 0; t < 8; ++t)
      #pragma unroll
      for (int v = 0; v < 8; ++v)
        ascr[(v + 8 * h) * 128 + t * 16 + ln] = (__bf16)OT[t][v];
    #pragma unroll
    for (int kc = 0; kc < 4; ++kc)
      Ach[kc] = ld_a16(ascr + ln * 128 + kc * 32 + h * 16);

    v8f DT[2];
    #pragma unroll
    for (int tn = 0; tn < 2; ++tn) {
      v8f c = {0.f, 0.f, 0.f, 0.f, 0.f, 0.f, 0.f, 0.f};
      #pragma unroll
      for (int kc = 0; kc < 4; ++kc) {
        v16bf b = ld_contig16(sCb + (tn * 16 + ln) * 128 + kc * 32 + h * 16);
        c = __builtin_amdgcn_wmma_f32_16x16x32_bf16(false, Ach[kc], false, b,
                                                    (short)0, c, false, false);
      }
      DT[tn] = c;
    }

    // --- VQ distances + graph bias + argmin over 32 codes ---
    int bidx[8];
    #pragma unroll
    for (int v = 0; v < 8; ++v) {
      float d0 = fsq[v] + cbsqr[0] - 2.f * DT[0][v];
      float d1 = fsq[v] + cbsqr[1] - 2.f * DT[1][v];
      if (step > 0) {
        d0 -= sSadj[sym[v] * 32 + ln];
        d1 -= sSadj[sym[v] * 32 + 16 + ln];
      }
      float best = d0; int bi = ln;
      if (d1 < best) { best = d1; bi = 16 + ln; }
      #pragma unroll
      for (int m = 1; m < 16; m <<= 1) {
        float ob = __shfl_xor(best, m, 32);
        int   oi = __shfl_xor(bi,   m, 32);
        if (ob < best || (ob == best && oi < bi)) { best = ob; bi = oi; }
      }
      bidx[v] = bi;
    }

    // --- ACT bookkeeping + straight-through state update ---
    float still[8], peff[8];
    #pragma unroll
    for (int v = 0; v < 8; ++v) {
      still[v] = (halting[v] < 0.99f) ? 1.f : 0.f;
      peff[v]  = (step == DEPTH - 1) ? remain[v] : p[v] * still[v];
    }
    #pragma unroll
    for (int t = 0; t < 8; ++t)
      #pragma unroll
      for (int v = 0; v < 8; ++v) {
        float zq = (float)sCb[bidx[v] * 128 + t * 16 + ln];
        float f  = OT[t][v];
        float df = zq - f;
        vqacc += df * df;
        float zn = 0.7f * f + 0.3f * zq;
        ST[t][v] = zn;
        AT[t][v] += peff[v] * zn;
      }
    #pragma unroll
    for (int v = 0; v < 8; ++v) {
      halting[v] += peff[v];
      remain[v]  -= peff[v];
      sym[v] = bidx[v];
    }
    if (ln == 0) {
      #pragma unroll
      for (int v = 0; v < 8; ++v) pacc += still[v];
    }
  }

  // --- decoder: [ar ai] @ dec_w.T + dec_b ---
  #pragma unroll
  for (int t = 0; t < 8; ++t)
    #pragma unroll
    for (int v = 0; v < 8; ++v)
      ascr[(v + 8 * h) * 128 + t * 16 + ln] = (__bf16)AT[t][v];
  v16bf Ach[4];
  #pragma unroll
  for (int kc = 0; kc < 4; ++kc)
    Ach[kc] = ld_a16(ascr + ln * 128 + kc * 32 + h * 16);

  #pragma unroll
  for (int tn = 0; tn < 2; ++tn) {
    v8f c = {0.f, 0.f, 0.f, 0.f, 0.f, 0.f, 0.f, 0.f};
    #pragma unroll
    for (int kc = 0; kc < 4; ++kc) {
      v16bf b = ld_contig16(sDw + (tn * 16 + ln) * 128 + kc * 32 + h * 16);
      c = __builtin_amdgcn_wmma_f32_16x16x32_bf16(false, Ach[kc], false, b,
                                                  (short)0, c, false, false);
    }
    #pragma unroll
    for (int v = 0; v < 8; ++v) {
      int row = base + v + 8 * h;
      out[row * 32 + tn * 16 + ln] = c[v] + decb[tn];
    }
  }

  // --- scalar outputs (ponder, vq_total) ---
  float vq = vqacc, pp = pacc;
  #pragma unroll
  for (int m = 1; m < 32; m <<= 1) {
    vq += __shfl_xor(vq, m, 32);
    pp += __shfl_xor(pp, m, 32);
  }
  if (lane == 0) {
    atomicAdd(out + BATCH * 32,     pp * (1.f / (float)BATCH));
    atomicAdd(out + BATCH * 32 + 1, vq * (1.25f / ((float)BATCH * 128.f)));
  }
}

extern "C" void kernel_launch(void* const* d_in, const int* in_sizes, int n_in,
                              void* d_out, int out_size, void* d_ws, size_t ws_size,
                              hipStream_t stream) {
  const int*   x         = (const int*)  d_in[0];
  const float* emb_mag   = (const float*)d_in[1];
  const float* emb_phase = (const float*)d_in[2];
  const float* Wr        = (const float*)d_in[3];
  const float* Wi        = (const float*)d_in[4];
  const float* ln_scale  = (const float*)d_in[5];
  const float* ln_shift  = (const float*)d_in[6];
  const float* mr_bias   = (const float*)d_in[7];
  const float* halt_w    = (const float*)d_in[8];
  const float* halt_b    = (const float*)d_in[9];
  const float* codebook  = (const float*)d_in[10];
  const float* adj       = (const float*)d_in[11];
  const float* dec_w     = (const float*)d_in[12];
  const float* dec_b     = (const float*)d_in[13];
  float* out = (float*)d_out;

  crsn_prep<<<1, 256, 0, stream>>>(Wr, Wi, codebook, adj, dec_w,
                                   out + (size_t)BATCH * 32, d_ws);
  crsn_main<<<BATCH / 128, 256, 0, stream>>>(x, emb_mag, emb_phase, ln_scale,
                                             ln_shift, mr_bias, halt_w, halt_b,
                                             dec_b, d_ws, out);
}
